// CBR_RNN_1864015806862
// MI455X (gfx1250) — compile-verified, matrix operand order checked by reading, not run on hardware
//
#include <hip/hip_runtime.h>
#include <hip/hip_bf16.h>

using bf16 = __bf16;
typedef __attribute__((ext_vector_type(16))) __bf16 v16bf;
typedef __attribute__((ext_vector_type(8)))  __bf16 bf16x8;
typedef __attribute__((ext_vector_type(8)))  float  v8f;

union AFrag { bf16x8 h[2]; v16bf v; };

static constexpr int S = 128, B = 32, V = 32000, NINP = 1024, NHID = 1024, NHEADS = 16, HD = 64;

// ---------------------------------------------------------------------------
// Weight retile: W is [N,K] fp32 row-major (used as x @ W^T). Pack bf16 into
// WMMA B-fragment blobs: blob(t=n_tile, c=k_chunk) holds 512 bf16, element
// index = lane*16 + j, matching the 16-bit fragment lane layout:
//   lane<16 : n = n0+lane, k in {k0..k0+7, k0+16..k0+23}
//   lane>=16: n = n0+lane-16, k in {k0+8..k0+15, k0+24..k0+31}
// ---------------------------------------------------------------------------
__global__ void tile_weights_kernel(const float* __restrict__ W, bf16* __restrict__ out,
                                    int N, int K) {
    long long idx = (long long)blockIdx.x * blockDim.x + threadIdx.x;
    long long total = (long long)N * K;
    if (idx >= total) return;
    int KC = K >> 5;
    long long blob = idx >> 9;
    int e = (int)(idx & 511);
    int l = e >> 4, j = e & 15;
    int t = (int)(blob / KC), c = (int)(blob % KC);
    int n = t * 16 + (l & 15);
    int k = c * 32 + ((l < 16) ? 0 : 8) + ((j < 8) ? j : j + 8);
    out[idx] = (bf16)W[(long long)n * K + k];
}

// Embedding gather + fp32 -> bf16
__global__ void embed_kernel(const int* __restrict__ obs, const float* __restrict__ enc_w,
                             bf16* __restrict__ emb, int total) {
    int idx = blockIdx.x * blockDim.x + threadIdx.x;
    if (idx >= total) return;
    int sb = idx / NINP, c = idx % NINP;
    int tok = obs[sb];
    emb[idx] = (bf16)enc_w[(long long)tok * NINP + c];
}

// Initialize caches, hidden slots, states_full[0]
__global__ void init_kernel(const float* __restrict__ hidden_init,
                            const float* __restrict__ kc_init, const float* __restrict__ vc_init,
                            float* __restrict__ key_cache, float* __restrict__ val_cache,
                            float* __restrict__ states0, bf16* __restrict__ xq_h, bf16* __restrict__ xi_h) {
    int idx = blockIdx.x * blockDim.x + threadIdx.x;
    if (idx >= B * NHID) return;
    key_cache[idx] = kc_init[idx];
    val_cache[idx] = vc_init[idx];
    float h = hidden_init[idx];
    states0[idx] = h;
    int b = idx / NHID, c = idx % NHID;
    xq_h[b * 2048 + c] = (bf16)h;   // xq cols [1024:2048) (base pre-offset)
    xi_h[b * 4096 + c] = (bf16)h;   // xi cols [3072:4096) (base pre-offset)
}

// Per-step: copy e_s into xq[:, :1024] and xi[:, :1024]
__global__ void pack_e_kernel(const bf16* __restrict__ emb_s, bf16* __restrict__ xq,
                              bf16* __restrict__ xi) {
    int idx = blockIdx.x * blockDim.x + threadIdx.x;
    if (idx >= B * NINP) return;
    int b = idx / NINP, c = idx % NINP;
    bf16 v = emb_s[idx];
    xq[b * 2048 + c] = v;
    xi[b * 4096 + c] = v;
}

// ---------------------------------------------------------------------------
// bf16 WMMA GEMM, 2x2 register-blocked: each wave owns a 32x32 macro-tile.
// Per K-chunk: 2 A frags + 2 B frags -> 4 WMMAs into 4 independent
// accumulators (2x data reuse per fragment, 4 interleaved XDL chains).
// A bf16 row-major; Bt pre-tiled 512-bf16 fragment blobs.
// ---------------------------------------------------------------------------
__global__ void wmma_gemm_kernel(const bf16* __restrict__ A, const bf16* __restrict__ Bt,
                                 float* __restrict__ C, const float* __restrict__ bias,
                                 int M, int K, int N) {
    const int wpb  = blockDim.x >> 5;
    const int wid  = blockIdx.x * wpb + (threadIdx.x >> 5);
    const int lane = threadIdx.x & 31;
    const int hl   = lane & 15;
    const int hi8  = (lane >> 4) * 8;
    const int mt2 = M >> 5, nt2 = N >> 5, kc = K >> 5;
    const int ntiles = mt2 * nt2;
    const int nwaves = gridDim.x * wpb;

    for (int t = wid; t < ntiles; t += nwaves) {
        const int tm2 = t % mt2, tn2 = t / mt2;
        const int m0 = tm2 << 5, n0 = tn2 << 5;
        v8f acc00 = {0.f, 0.f, 0.f, 0.f, 0.f, 0.f, 0.f, 0.f};
        v8f acc01 = acc00, acc10 = acc00, acc11 = acc00;
        const bf16* arow0  = A + (long long)(m0 + hl) * K;
        const bf16* arow1  = A + (long long)(m0 + 16 + hl) * K;
        const bf16* bblob0 = Bt + (long long)(2 * tn2)     * kc * 512 + lane * 16;
        const bf16* bblob1 = Bt + (long long)(2 * tn2 + 1) * kc * 512 + lane * 16;
        for (int c = 0; c < kc; ++c) {
            const int k0 = c << 5;
            AFrag a0, a1, b0, b1;
            a0.h[0] = *reinterpret_cast<const bf16x8*>(arow0 + k0 + hi8);
            a0.h[1] = *reinterpret_cast<const bf16x8*>(arow0 + k0 + hi8 + 16);
            a1.h[0] = *reinterpret_cast<const bf16x8*>(arow1 + k0 + hi8);
            a1.h[1] = *reinterpret_cast<const bf16x8*>(arow1 + k0 + hi8 + 16);
            const bf16x8* bp0 = reinterpret_cast<const bf16x8*>(bblob0 + (long long)c * 512);
            const bf16x8* bp1 = reinterpret_cast<const bf16x8*>(bblob1 + (long long)c * 512);
            b0.h[0] = bp0[0]; b0.h[1] = bp0[1];
            b1.h[0] = bp1[0]; b1.h[1] = bp1[1];
            __builtin_prefetch((const void*)(bblob0 + (long long)(c + 1) * 512), 0, 0);
            __builtin_prefetch((const void*)(bblob1 + (long long)(c + 1) * 512), 0, 0);
            acc00 = __builtin_amdgcn_wmma_f32_16x16x32_bf16(
                false, a0.v, false, b0.v, (short)0, acc00, false, false);
            acc01 = __builtin_amdgcn_wmma_f32_16x16x32_bf16(
                false, a0.v, false, b1.v, (short)0, acc01, false, false);
            acc10 = __builtin_amdgcn_wmma_f32_16x16x32_bf16(
                false, a1.v, false, b0.v, (short)0, acc10, false, false);
            acc11 = __builtin_amdgcn_wmma_f32_16x16x32_bf16(
                false, a1.v, false, b1.v, (short)0, acc11, false, false);
        }
        const int col0 = n0 + hl, col1 = n0 + 16 + hl;
        const float bv0 = bias ? bias[col0] : 0.f;
        const float bv1 = bias ? bias[col1] : 0.f;
        float* cp0 = C + (long long)(m0 + hi8) * N;
        float* cp1 = C + (long long)(m0 + 16 + hi8) * N;
#pragma unroll
        for (int r = 0; r < 8; ++r) {
            cp0[(long long)r * N + col0] = acc00[r] + bv0;
            cp0[(long long)r * N + col1] = acc01[r] + bv1;
            cp1[(long long)r * N + col0] = acc10[r] + bv0;
            cp1[(long long)r * N + col1] = acc11[r] + bv1;
        }
    }
}

// ---------------------------------------------------------------------------
// Fused bias + LayerNorm + ReLU over each row of raw[M,N]; one block per row.
// Optional fp32 and bf16 destinations (with strides) for downstream consumers.
// ---------------------------------------------------------------------------
__global__ void ln_relu_kernel(const float* __restrict__ raw, const float* __restrict__ bias,
                               const float* __restrict__ gamma, const float* __restrict__ beta,
                               int N, float* outf, int ldf, bf16* outb, int ldb) {
    __shared__ float sh1[256], sh2[256];
    const int row = blockIdx.x;
    const float* r = raw + (long long)row * N;
    float s = 0.f, s2 = 0.f;
    for (int i = threadIdx.x; i < N; i += blockDim.x) {
        float v = r[i] + bias[i];
        s += v; s2 += v * v;
    }
    sh1[threadIdx.x] = s; sh2[threadIdx.x] = s2;
    __syncthreads();
    for (int off = blockDim.x >> 1; off > 0; off >>= 1) {
        if (threadIdx.x < off) {
            sh1[threadIdx.x] += sh1[threadIdx.x + off];
            sh2[threadIdx.x] += sh2[threadIdx.x + off];
        }
        __syncthreads();
    }
    const float mu   = sh1[0] / N;
    const float var  = sh2[0] / N - mu * mu;
    const float rstd = rsqrtf(var + 1e-5f);
    for (int i = threadIdx.x; i < N; i += blockDim.x) {
        float v = r[i] + bias[i];
        float y = (v - mu) * rstd * gamma[i] + beta[i];
        y = fmaxf(y, 0.f);
        if (outf) outf[(long long)row * ldf + i] = y;
        if (outb) outb[(long long)row * ldb + i] = (bf16)y;
    }
}

// ---------------------------------------------------------------------------
// Attention: one wave32 per (b, head). Online softmax over cache_len entries,
// each lane owns 2 of the 64 head dims. Writes bf16 into xi[:, 2048:3072).
// ---------------------------------------------------------------------------
__global__ void attn_kernel(const float* __restrict__ q, const float* __restrict__ keys,
                            const float* __restrict__ vals, int cache_len,
                            bf16* __restrict__ xo, int ldo) {
    const int wid  = (blockIdx.x * blockDim.x + threadIdx.x) >> 5;
    const int lane = threadIdx.x & 31;
    if (wid >= B * NHEADS) return;
    const int b = wid >> 4, h = wid & 15;
    const int base = h * HD + lane * 2;
    const float q0 = q[(long long)b * NHID + base];
    const float q1 = q[(long long)b * NHID + base + 1];
    const float scale = 0.125f;  // 1/sqrt(64)
    float m = -3.0e38f, ssum = 0.f, a0 = 0.f, a1 = 0.f;
    for (int c = 0; c < cache_len; ++c) {
        const long long off = ((long long)c * B + b) * NHID + base;
        float p = q0 * keys[off] + q1 * keys[off + 1];
#pragma unroll
        for (int o = 16; o > 0; o >>= 1) p += __shfl_xor(p, o, 32);
        p *= scale;
        const float mn   = fmaxf(m, p);
        const float corr = __expf(m - mn);
        const float w    = __expf(p - mn);
        a0 = a0 * corr + w * vals[off];
        a1 = a1 * corr + w * vals[off + 1];
        ssum = ssum * corr + w;
        m = mn;
    }
    const float inv = 1.0f / ssum;
    xo[(long long)b * ldo + base]     = (bf16)(a0 * inv);
    xo[(long long)b * ldo + base + 1] = (bf16)(a1 * inv);
}

// ---------------------------------------------------------------------------
// Split final[:, 0:1024|1024:2048|2048:3072] -> K-cache, V-cache, hidden state.
// Hidden also goes (bf16) into next step's xq/xi slots and decoder A buffer.
// ---------------------------------------------------------------------------
__global__ void scatter_fin_kernel(const float* __restrict__ fin,
                                   float* __restrict__ kc_row, float* __restrict__ vc_row,
                                   float* __restrict__ states_row,
                                   bf16* __restrict__ xq_h, bf16* __restrict__ xi_h,
                                   bf16* __restrict__ states_bf_row) {
    int idx = blockIdx.x * blockDim.x + threadIdx.x;
    if (idx >= B * 3 * NHID) return;
    int b = idx / (3 * NHID), c = idx % (3 * NHID);
    float v = fin[idx];
    if (c < NHID) {
        kc_row[b * NHID + c] = v;
    } else if (c < 2 * NHID) {
        vc_row[b * NHID + (c - NHID)] = v;
    } else {
        int cc = c - 2 * NHID;
        states_row[b * NHID + cc]   = v;
        xq_h[b * 2048 + cc]         = (bf16)v;
        xi_h[b * 4096 + cc]         = (bf16)v;
        states_bf_row[b * NHID + cc] = (bf16)v;
    }
}

// ---------------------------------------------------------------------------
extern "C" void kernel_launch(void* const* d_in, const int* in_sizes, int n_in,
                              void* d_out, int out_size, void* d_ws, size_t ws_size,
                              hipStream_t stream) {
    const int*   obs        = (const int*)  d_in[0];
    const float* hidden_ini = (const float*)d_in[1];
    const float* kc_ini     = (const float*)d_in[2];
    const float* vc_ini     = (const float*)d_in[3];
    const float* enc_w      = (const float*)d_in[4];
    const float* q_w        = (const float*)d_in[5];
    const float* q_b        = (const float*)d_in[6];
    const float* qn_g       = (const float*)d_in[7];
    const float* qn_b       = (const float*)d_in[8];
    const float* int_w      = (const float*)d_in[9];
    const float* int_b      = (const float*)d_in[10];
    const float* intn_g     = (const float*)d_in[11];
    const float* intn_b     = (const float*)d_in[12];
    const float* fin_w      = (const float*)d_in[13];
    const float* fin_b      = (const float*)d_in[14];
    const float* fn_g       = (const float*)d_in[15];
    const float* fn_b       = (const float*)d_in[16];
    const float* dec_w      = (const float*)d_in[17];
    const float* dec_b      = (const float*)d_in[18];

    float* logits     = (float*)d_out;                               // [S,B,V]
    float* states_out = (float*)d_out + (long long)S * B * V;        // [S+1,B,NHID]

    // ---- workspace carve-up (256B aligned) ----
    char* ws = (char*)d_ws;
    size_t off = 0;
    auto alloc = [&](size_t bytes) -> void* {
        void* p = ws + off;
        off += (bytes + 255) & ~(size_t)255;
        return p;
    };
    bf16* qw_t   = (bf16*)alloc((size_t)NHID * 2048 * sizeof(bf16));          // [1024,2048]
    bf16* iw_t   = (bf16*)alloc((size_t)4 * NHID * 4 * NHID * sizeof(bf16));  // [4096,4096]
    bf16* fw_t   = (bf16*)alloc((size_t)3 * NHID * 4 * NHID * sizeof(bf16));  // [3072,4096]
    bf16* dw_t   = (bf16*)alloc((size_t)V * NHID * sizeof(bf16));             // [32000,1024]
    bf16* emb    = (bf16*)alloc((size_t)S * B * NINP * sizeof(bf16));
    float* keyc  = (float*)alloc((size_t)(S + 1) * B * NHID * sizeof(float));
    float* valc  = (float*)alloc((size_t)(S + 1) * B * NHID * sizeof(float));
    bf16* xq     = (bf16*)alloc((size_t)B * 2048 * sizeof(bf16));
    bf16* xi     = (bf16*)alloc((size_t)B * 4096 * sizeof(bf16));
    float* qbuf  = (float*)alloc((size_t)B * NHID * sizeof(float));
    bf16* inter  = (bf16*)alloc((size_t)B * 4 * NHID * sizeof(bf16));
    float* raw   = (float*)alloc((size_t)B * 4 * NHID * sizeof(float));
    float* fino  = (float*)alloc((size_t)B * 3 * NHID * sizeof(float));
    bf16* st_bf  = (bf16*)alloc((size_t)S * B * NHID * sizeof(bf16));
    (void)ws_size; (void)in_sizes; (void)n_in; (void)out_size;

    auto gemm = [&](const bf16* A, const bf16* Bt, float* C, const float* bias,
                    int M, int K, int N) {
        int ntiles = (M >> 5) * (N >> 5);      // 32x32 macro-tiles
        int blocks = (ntiles + 7) / 8;         // 8 waves per 256-thread block
        wmma_gemm_kernel<<<dim3(blocks), dim3(256), 0, stream>>>(A, Bt, C, bias, M, K, N);
    };
    auto grid1d = [](long long total) { return dim3((unsigned)((total + 255) / 256)); };

    // ---- one-time preprocessing ----
    tile_weights_kernel<<<grid1d((long long)NHID * 2048), 256, 0, stream>>>(q_w, qw_t, NHID, 2048);
    tile_weights_kernel<<<grid1d((long long)4 * NHID * 4 * NHID), 256, 0, stream>>>(int_w, iw_t, 4 * NHID, 4 * NHID);
    tile_weights_kernel<<<grid1d((long long)3 * NHID * 4 * NHID), 256, 0, stream>>>(fin_w, fw_t, 3 * NHID, 4 * NHID);
    tile_weights_kernel<<<grid1d((long long)V * NHID), 256, 0, stream>>>(dec_w, dw_t, V, NHID);
    embed_kernel<<<grid1d((long long)S * B * NINP), 256, 0, stream>>>(obs, enc_w, emb, S * B * NINP);
    init_kernel<<<grid1d(B * NHID), 256, 0, stream>>>(hidden_ini, kc_ini, vc_ini,
                                                      keyc, valc, states_out,
                                                      xq + 1024, xi + 3072);

    // ---- recurrent steps ----
    for (int s = 0; s < S; ++s) {
        pack_e_kernel<<<grid1d(B * NINP), 256, 0, stream>>>(emb + (long long)s * B * NINP, xq, xi);

        // q = relu(ln(concat(e,h) @ q_w.T + q_b))
        gemm(xq, qw_t, raw, nullptr, B, 2048, NHID);
        ln_relu_kernel<<<B, 256, 0, stream>>>(raw, q_b, qn_g, qn_b, NHID,
                                              qbuf, NHID, xi + 1024, 4096);

        // multi-head attention over cache rows [0..s]
        attn_kernel<<<dim3((B * NHEADS * 32 + 255) / 256), 256, 0, stream>>>(
            qbuf, keyc, valc, s + 1, xi + 2048, 4096);

        // inter = relu(ln(concat(e,q,attn,h) @ int_w.T + int_b))
        gemm(xi, iw_t, raw, nullptr, B, 4 * NHID, 4 * NHID);
        ln_relu_kernel<<<B, 256, 0, stream>>>(raw, int_b, intn_g, intn_b, 4 * NHID,
                                              nullptr, 0, inter, 4 * NHID);

        // final = relu(ln(inter @ fin_w.T + fin_b))
        gemm(inter, fw_t, raw, nullptr, B, 4 * NHID, 3 * NHID);
        ln_relu_kernel<<<B, 256, 0, stream>>>(raw, fin_b, fn_g, fn_b, 3 * NHID,
                                              fino, 3 * NHID, nullptr, 0);

        scatter_fin_kernel<<<grid1d((long long)B * 3 * NHID), 256, 0, stream>>>(
            fino,
            keyc + (long long)(s + 1) * B * NHID,
            valc + (long long)(s + 1) * B * NHID,
            states_out + (long long)(s + 1) * B * NHID,
            xq + 1024, xi + 3072,
            st_bf + (long long)s * B * NHID);
    }

    // ---- decoder: logits = states @ dec_w.T + dec_b  (4096 x 32000, K=1024) ----
    gemm(st_bf, dw_t, logits, dec_b, S * B, NHID, V);
}
